// StandardAttention_15049565405218
// MI455X (gfx1250) — matrix operand order repaired and checked by reading, not compiled, not run on hardware
//
#include <hip/hip_runtime.h>
#include <hip/hip_bf16.h>
#include <math.h>

// ---------------------------------------------------------------------------
// Types for CDNA5 WMMA (gfx1250, wave32)
// ---------------------------------------------------------------------------
typedef __attribute__((ext_vector_type(16))) __bf16 v16bf;
typedef __attribute__((ext_vector_type(8)))  __bf16 v8bf;
typedef __attribute__((ext_vector_type(8)))  float  v8f;

#define D_MODEL 1024
#define N_HEADS 16
#define D_HEAD  64
#define BATCH   2
#define SEQ     2048
#define MTOT    (BATCH * SEQ)          // 4096 rows for the big GEMMs

__device__ __forceinline__ v8f wmma_bf16(v16bf a, v16bf b, v8f c) {
    return __builtin_amdgcn_wmma_f32_16x16x32_bf16(
        /*neg_a=*/false, a, /*neg_b=*/false, b,
        /*c_mod=*/(short)0, c, /*reuse_a=*/false, /*reuse_b=*/false);
}

__device__ __forceinline__ v16bf cat16(v8bf lo, v8bf hi) {
    return __builtin_shufflevector(lo, hi, 0, 1, 2, 3, 4, 5, 6, 7,
                                           8, 9, 10, 11, 12, 13, 14, 15);
}

// ---------------------------------------------------------------------------
// Fragment loaders (CDNA5 ISA layouts, wave32).
// A (16x32, row-major src): lane L<16 row M=L holds K 0..7 / 16..23,
// lane L+16 holds K 8..15 / 24..31 -> two contiguous 16B runs per lane.
// ---------------------------------------------------------------------------
__device__ __forceinline__ v16bf load_a_frag(const __bf16* A, int lda,
                                             int m0, int k0, int lane) {
    int row = m0 + (lane & 15);
    int kb  = k0 + ((lane >> 4) << 3);          // +0 or +8
    const __bf16* p = A + (long)row * lda + kb;
    v8bf lo = *(const v8bf*)(p);
    v8bf hi = *(const v8bf*)(p + 16);
    return cat16(lo, hi);
}

// B (32x16) where B(k,n) = T[n][k]; T row-major, leading dim ldt.
// Same K order as A: lanes 0-15 hold K 0..7 / 16..23, lanes 16-31 K 8..15 / 24..31
// -> two contiguous 16B runs per lane.
__device__ __forceinline__ v16bf load_b_frag_tr(const __bf16* T, int ldt,
                                                int k0, int n0, int lane) {
    int col = n0 + (lane & 15);
    int kb  = k0 + ((lane >> 4) << 3);          // +0 or +8
    const __bf16* p = T + (long)col * ldt + kb;
    v8bf lo = *(const v8bf*)(p);
    v8bf hi = *(const v8bf*)(p + 16);
    return cat16(lo, hi);
}

// Same, with the column (n) clamped (for K rows past seq end; masked anyway).
__device__ __forceinline__ v16bf load_b_frag_tr_colclamp(const __bf16* T, int ldt,
                                                         int k0, int n0, int lane,
                                                         int maxCol) {
    int col = n0 + (lane & 15); if (col > maxCol) col = maxCol;
    int kb  = k0 + ((lane >> 4) << 3);
    const __bf16* p = T + (long)col * ldt + kb;
    v8bf lo = *(const v8bf*)(p);
    v8bf hi = *(const v8bf*)(p + 16);
    return cat16(lo, hi);
}

// Same, with the K-block start clamped so loads stay in-bounds
// (for V keys past seq end; their P weights are exactly 0).
__device__ __forceinline__ v16bf load_b_frag_tr_kclamp(const __bf16* T, int ldt,
                                                       int k0, int n0, int lane,
                                                       int maxK0) {
    int col = n0 + (lane & 15);
    int kb  = (k0 > maxK0 - 16 ? maxK0 - 16 : k0) + ((lane >> 4) << 3);
    const __bf16* p = T + (long)col * ldt + kb;
    v8bf lo = *(const v8bf*)(p);
    v8bf hi = *(const v8bf*)(p + 16);
    return cat16(lo, hi);
}

// ---------------------------------------------------------------------------
// fp32 -> bf16 conversion (plain) and weight transpose
// ---------------------------------------------------------------------------
__global__ void cvt_f32_bf16(const float* __restrict__ src,
                             __bf16* __restrict__ dst, int n) {
    int i = blockIdx.x * blockDim.x + threadIdx.x;
    if (i < n) dst[i] = (__bf16)src[i];
}

// W [1024x1024] fp32 row-major -> Wt [1024x1024] bf16 with Wt[n][k] = W[k][n]
__global__ void cvt_transpose_w(const float* __restrict__ src,
                                __bf16* __restrict__ dst) {
    int idx = blockIdx.x * blockDim.x + threadIdx.x;   // over 2^20
    int k = idx >> 10, n = idx & 1023;
    dst[(long)n * D_MODEL + k] = (__bf16)src[idx];
}

// ---------------------------------------------------------------------------
// Projection GEMM: one wave computes a 16(M) x 64(N) output strip, with a
// software-pipelined K loop. Relaxed occupancy (waves_per_eu=1) raises the
// VGPR budget to 256 so the double-buffered fragments live in registers.
// mode 0: out[b][h][s][dh]   (Q and K)
// mode 1: out[b][h][dh][s]   (V, stored transposed for P@V fragments)
// ---------------------------------------------------------------------------
__global__ void
__launch_bounds__(128, 1)
__attribute__((amdgpu_waves_per_eu(1)))
proj_kernel(const __bf16* __restrict__ X,
            const __bf16* __restrict__ Wt,
            const float*  __restrict__ bias,
            __bf16* __restrict__ out, float scale, int mode) {
    int lane = threadIdx.x & 31;
    int wid  = blockIdx.x * (blockDim.x >> 5) + (threadIdx.x >> 5);
    int tm  = wid >> 4;                // 0..255  (4096/16)
    int tn4 = wid & 15;                // 0..15   (1024/64)
    int m0 = tm << 4, n0 = tn4 << 6;

    v8f c0 = {}, c1 = {}, c2 = {}, c3 = {};

    v16bf a  = load_a_frag(X, D_MODEL, m0, 0, lane);
    v16bf b0 = load_b_frag_tr(Wt, D_MODEL, 0, n0 +  0, lane);
    v16bf b1 = load_b_frag_tr(Wt, D_MODEL, 0, n0 + 16, lane);
    v16bf b2 = load_b_frag_tr(Wt, D_MODEL, 0, n0 + 32, lane);
    v16bf b3 = load_b_frag_tr(Wt, D_MODEL, 0, n0 + 48, lane);

    for (int k0 = 0; k0 < D_MODEL - 32; k0 += 32) {
        int kn = k0 + 32;
        v16bf an  = load_a_frag(X, D_MODEL, m0, kn, lane);
        v16bf bn0 = load_b_frag_tr(Wt, D_MODEL, kn, n0 +  0, lane);
        v16bf bn1 = load_b_frag_tr(Wt, D_MODEL, kn, n0 + 16, lane);
        v16bf bn2 = load_b_frag_tr(Wt, D_MODEL, kn, n0 + 32, lane);
        v16bf bn3 = load_b_frag_tr(Wt, D_MODEL, kn, n0 + 48, lane);
        c0 = wmma_bf16(a, b0, c0);
        c1 = wmma_bf16(a, b1, c1);
        c2 = wmma_bf16(a, b2, c2);
        c3 = wmma_bf16(a, b3, c3);
        a = an; b0 = bn0; b1 = bn1; b2 = bn2; b3 = bn3;
    }
    c0 = wmma_bf16(a, b0, c0);
    c1 = wmma_bf16(a, b1, c1);
    c2 = wmma_bf16(a, b2, c2);
    c3 = wmma_bf16(a, b3, c3);

    int c  = lane & 15;
    int mb = (lane >> 4) << 3;
#pragma unroll
    for (int t = 0; t < 4; ++t) {
        v8f acc = (t == 0) ? c0 : (t == 1) ? c1 : (t == 2) ? c2 : c3;
        int col = n0 + (t << 4) + c;
        float bv = bias[col];
        int h  = col >> 6;
        int dh = col & 63;
#pragma unroll
        for (int i = 0; i < 8; ++i) {
            int m  = m0 + mb + i;
            int bb = m >> 11;
            int s  = m & (SEQ - 1);
            float v = (acc[i] + bv) * scale;
            long idx = (mode == 0)
                ? (((long)(bb * N_HEADS + h) * SEQ) + s) * D_HEAD + dh
                : (((long)(bb * N_HEADS + h) * D_HEAD) + dh) * SEQ + s;
            out[idx] = (__bf16)v;
        }
    }
}

// ---------------------------------------------------------------------------
// Causal flash attention per (b,h, 16-query tile). One wave per block.
// Q pre-scaled by 1/sqrt(Dh). Key blocks of 32 -> P@V has WMMA K=32.
// Q,K: [B,H,S,64]; V: [B,H,64,S] (transposed); O: [B,S,1024] bf16.
// V fragments are loaded BEFORE the softmax so their latency hides behind it.
// ---------------------------------------------------------------------------
__global__ void
__launch_bounds__(32)
__attribute__((amdgpu_waves_per_eu(2)))
attn_kernel(const __bf16* __restrict__ Q,
            const __bf16* __restrict__ K,
            const __bf16* __restrict__ Vt,
            __bf16* __restrict__ O) {
    __shared__ __align__(16) __bf16 Plds[16 * 32];   // one wave per block

    int lane = threadIdx.x & 31;
    int g  = blockIdx.x;               // 0 .. B*H*(S/16)-1 = 4095
    int bh = g >> 7;
    int qt = g & 127;
    int q0 = qt << 4;

    const __bf16* Qb = Q  + (long)bh * SEQ * D_HEAD;
    const __bf16* Kb = K  + (long)bh * SEQ * D_HEAD;
    const __bf16* Vb = Vt + (long)bh * SEQ * D_HEAD;

    // Q fragments are loop-invariant: hoist.
    v16bf aq0 = load_a_frag(Qb, D_HEAD, q0, 0,  lane);
    v16bf aq1 = load_a_frag(Qb, D_HEAD, q0, 32, lane);

    v8f o0 = {}, o1 = {}, o2 = {}, o3 = {};
    float mrow[8], lrow[8];
#pragma unroll
    for (int i = 0; i < 8; ++i) { mrow[i] = -INFINITY; lrow[i] = 0.f; }

    int c  = lane & 15;
    int mb = (lane >> 4) << 3;

    for (int j0 = 0; j0 <= q0 + 15; j0 += 32) {
        // ---- scores: 16x32 via two 16x16 WMMA tiles, Dh in 2 K-steps
        v8f s0 = {}, s1 = {};
        {
            v16bf b00 = load_b_frag_tr_colclamp(Kb, D_HEAD, 0,  j0,      lane, SEQ - 1);
            v16bf b01 = load_b_frag_tr_colclamp(Kb, D_HEAD, 0,  j0 + 16, lane, SEQ - 1);
            v16bf b10 = load_b_frag_tr_colclamp(Kb, D_HEAD, 32, j0,      lane, SEQ - 1);
            v16bf b11 = load_b_frag_tr_colclamp(Kb, D_HEAD, 32, j0 + 16, lane, SEQ - 1);
            s0 = wmma_bf16(aq0, b00, s0);
            s1 = wmma_bf16(aq0, b01, s1);
            s0 = wmma_bf16(aq1, b10, s0);
            s1 = wmma_bf16(aq1, b11, s1);
        }

        // ---- V fragments in flight while we do the softmax VALU work
        v16bf bv0 = load_b_frag_tr_kclamp(Vb, SEQ, j0, 0,  lane, SEQ - 16);
        v16bf bv1 = load_b_frag_tr_kclamp(Vb, SEQ, j0, 16, lane, SEQ - 16);
        v16bf bv2 = load_b_frag_tr_kclamp(Vb, SEQ, j0, 32, lane, SEQ - 16);
        v16bf bv3 = load_b_frag_tr_kclamp(Vb, SEQ, j0, 48, lane, SEQ - 16);

        // ---- causal mask
        int col0 = j0 + c, col1 = col0 + 16;
#pragma unroll
        for (int i = 0; i < 8; ++i) {
            int row = q0 + mb + i;
            if (col0 > row) s0[i] = -1e30f;
            if (col1 > row) s1[i] = -1e30f;
        }

        // ---- online softmax (row lives in one half-wave; reduce 16 lanes)
        float scl[8];
#pragma unroll
        for (int i = 0; i < 8; ++i) {
            float v = fmaxf(s0[i], s1[i]);
            v = fmaxf(v, __shfl_xor(v, 1));
            v = fmaxf(v, __shfl_xor(v, 2));
            v = fmaxf(v, __shfl_xor(v, 4));
            v = fmaxf(v, __shfl_xor(v, 8));
            float nm = fmaxf(mrow[i], v);
            scl[i] = __expf(mrow[i] - nm);
            mrow[i] = nm;
            float p0 = __expf(s0[i] - nm);
            float p1 = __expf(s1[i] - nm);
            s0[i] = p0; s1[i] = p1;
            float rs = p0 + p1;
            rs += __shfl_xor(rs, 1);
            rs += __shfl_xor(rs, 2);
            rs += __shfl_xor(rs, 4);
            rs += __shfl_xor(rs, 8);
            lrow[i] = lrow[i] * scl[i] + rs;
        }
#pragma unroll
        for (int i = 0; i < 8; ++i) {
            o0[i] *= scl[i]; o1[i] *= scl[i]; o2[i] *= scl[i]; o3[i] *= scl[i];
        }

        // ---- stage P (16x32) to LDS row-major, reload as A fragment
#pragma unroll
        for (int i = 0; i < 8; ++i) {
            Plds[(mb + i) * 32 + c]      = (__bf16)s0[i];
            Plds[(mb + i) * 32 + 16 + c] = (__bf16)s1[i];
        }
        asm volatile("s_wait_dscnt 0x0" ::: "memory");
        v16bf ap = load_a_frag((const __bf16*)Plds, 32, 0, 0, lane);

        // ---- O += P @ V
        o0 = wmma_bf16(ap, bv0, o0);
        o1 = wmma_bf16(ap, bv1, o1);
        o2 = wmma_bf16(ap, bv2, o2);
        o3 = wmma_bf16(ap, bv3, o3);
        asm volatile("s_wait_dscnt 0x0" ::: "memory");
    }

    // ---- finalize: divide by l, store bf16 into [B,S,D], col = h*64+dh
    int bb = bh >> 4;
    int h  = bh & 15;
#pragma unroll
    for (int i = 0; i < 8; ++i) {
        float inv = 1.0f / lrow[i];
        int srow = q0 + mb + i;
        long base = ((long)bb * SEQ + srow) * D_MODEL + h * D_HEAD;
        O[base +  0 + c] = (__bf16)(o0[i] * inv);
        O[base + 16 + c] = (__bf16)(o1[i] * inv);
        O[base + 32 + c] = (__bf16)(o2[i] * inv);
        O[base + 48 + c] = (__bf16)(o3[i] * inv);
    }
}

// ---------------------------------------------------------------------------
// Output projection: out(f32 [4096x1024]) = A(bf16) @ Wo, Wo given transposed.
// One wave per 16x64 strip; software-pipelined K loop, relaxed occupancy.
// ---------------------------------------------------------------------------
__global__ void
__launch_bounds__(128, 1)
__attribute__((amdgpu_waves_per_eu(1)))
oproj_kernel(const __bf16* __restrict__ A,
             const __bf16* __restrict__ Wt,
             float* __restrict__ out) {
    int lane = threadIdx.x & 31;
    int wid  = blockIdx.x * (blockDim.x >> 5) + (threadIdx.x >> 5);
    int tm  = wid >> 4;
    int tn4 = wid & 15;
    int m0 = tm << 4, n0 = tn4 << 6;

    v8f c0 = {}, c1 = {}, c2 = {}, c3 = {};

    v16bf a  = load_a_frag(A, D_MODEL, m0, 0, lane);
    v16bf b0 = load_b_frag_tr(Wt, D_MODEL, 0, n0 +  0, lane);
    v16bf b1 = load_b_frag_tr(Wt, D_MODEL, 0, n0 + 16, lane);
    v16bf b2 = load_b_frag_tr(Wt, D_MODEL, 0, n0 + 32, lane);
    v16bf b3 = load_b_frag_tr(Wt, D_MODEL, 0, n0 + 48, lane);

    for (int k0 = 0; k0 < D_MODEL - 32; k0 += 32) {
        int kn = k0 + 32;
        v16bf an  = load_a_frag(A, D_MODEL, m0, kn, lane);
        v16bf bn0 = load_b_frag_tr(Wt, D_MODEL, kn, n0 +  0, lane);
        v16bf bn1 = load_b_frag_tr(Wt, D_MODEL, kn, n0 + 16, lane);
        v16bf bn2 = load_b_frag_tr(Wt, D_MODEL, kn, n0 + 32, lane);
        v16bf bn3 = load_b_frag_tr(Wt, D_MODEL, kn, n0 + 48, lane);
        c0 = wmma_bf16(a, b0, c0);
        c1 = wmma_bf16(a, b1, c1);
        c2 = wmma_bf16(a, b2, c2);
        c3 = wmma_bf16(a, b3, c3);
        a = an; b0 = bn0; b1 = bn1; b2 = bn2; b3 = bn3;
    }
    c0 = wmma_bf16(a, b0, c0);
    c1 = wmma_bf16(a, b1, c1);
    c2 = wmma_bf16(a, b2, c2);
    c3 = wmma_bf16(a, b3, c3);

    int c  = lane & 15;
    int mb = (lane >> 4) << 3;
#pragma unroll
    for (int t = 0; t < 4; ++t) {
        v8f acc = (t == 0) ? c0 : (t == 1) ? c1 : (t == 2) ? c2 : c3;
        int col = n0 + (t << 4) + c;
#pragma unroll
        for (int i = 0; i < 8; ++i) {
            int m = m0 + mb + i;
            out[(long)m * D_MODEL + col] = acc[i];
        }
    }
}

// ---------------------------------------------------------------------------
// Host-side launch
// ---------------------------------------------------------------------------
extern "C" void kernel_launch(void* const* d_in, const int* in_sizes, int n_in,
                              void* d_out, int out_size, void* d_ws, size_t ws_size,
                              hipStream_t stream) {
    const float* x  = (const float*)d_in[0];
    const float* Wq = (const float*)d_in[1];
    const float* bq = (const float*)d_in[2];
    const float* Wk = (const float*)d_in[3];
    const float* bk = (const float*)d_in[4];
    const float* Wv = (const float*)d_in[5];
    const float* bv = (const float*)d_in[6];
    const float* Wo = (const float*)d_in[7];
    float* out = (float*)d_out;

    // workspace layout (bytes)
    char* ws = (char*)d_ws;
    const size_t SZ_X = (size_t)MTOT * D_MODEL * 2;       // 8 MiB
    const size_t SZ_W = (size_t)D_MODEL * D_MODEL * 2;    // 2 MiB
    __bf16* xb  = (__bf16*)(ws);
    __bf16* wqt = (__bf16*)(ws + SZ_X);
    __bf16* wkt = (__bf16*)(ws + SZ_X + SZ_W);
    __bf16* wvt = (__bf16*)(ws + SZ_X + 2 * SZ_W);
    __bf16* wot = (__bf16*)(ws + SZ_X + 3 * SZ_W);
    __bf16* Qb  = (__bf16*)(ws + SZ_X + 4 * SZ_W);
    __bf16* Kb  = (__bf16*)(ws + 2 * SZ_X + 4 * SZ_W);
    __bf16* Vtb = (__bf16*)(ws + 3 * SZ_X + 4 * SZ_W);
    __bf16* Ob  = (__bf16*)(ws + 4 * SZ_X + 4 * SZ_W);

    // 1) fp32 -> bf16 conversions; weights transposed for contiguous B frags
    int nX = MTOT * D_MODEL;        // 4194304
    int nW = D_MODEL * D_MODEL;     // 1048576
    cvt_f32_bf16<<<(nX + 255) / 256, 256, 0, stream>>>(x, xb, nX);
    cvt_transpose_w<<<nW / 256, 256, 0, stream>>>(Wq, wqt);
    cvt_transpose_w<<<nW / 256, 256, 0, stream>>>(Wk, wkt);
    cvt_transpose_w<<<nW / 256, 256, 0, stream>>>(Wv, wvt);
    cvt_transpose_w<<<nW / 256, 256, 0, stream>>>(Wo, wot);

    // 2) projections: 4096 waves (16x64 strips), 4 waves/block
    const float qscale = 0.125f;    // 1/sqrt(64)
    proj_kernel<<<1024, 128, 0, stream>>>(xb, wqt, bq, Qb,  qscale, 0);
    proj_kernel<<<1024, 128, 0, stream>>>(xb, wkt, bk, Kb,  1.0f,   0);
    proj_kernel<<<1024, 128, 0, stream>>>(xb, wvt, bv, Vtb, 1.0f,   1);

    // 3) attention: B*H*(S/16) = 4096 waves, one per block
    attn_kernel<<<4096, 32, 0, stream>>>(Qb, Kb, Vtb, Ob);

    // 4) output projection -> fp32 d_out
    oproj_kernel<<<1024, 128, 0, stream>>>(Ob, wot, out);
}
